// WaveletUp_36318243455521
// MI455X (gfx1250) — matrix operand order repaired and checked by reading, not compiled, not run on hardware
//
#include <hip/hip_runtime.h>
#include <hip/hip_bf16.h>

typedef __attribute__((ext_vector_type(16))) _Float16 v16h;
typedef __attribute__((ext_vector_type(8)))  _Float16 v8h;
typedef __attribute__((ext_vector_type(8)))  float    v8f;

// ---------------- workspace layout (bytes) ----------------
// hp1  : [16][66][66][512] f16  padded NHWC concat buffer (skip ch 0..255, image 256..511)
// hp2  : [16][66][66][256] f16  padded NHWC conv1 output after BN+ReLU
// stats: 1024 f32 (sum1, sq1, sum2, sq2)
// xh   : [16][32][32][512] f16  x transposed to NHWC
// Whp/W1p/W2p : weights packed into WMMA A-fragment lane layout
// raw1 : [16][64][64][256] f16  conv1 raw output (NHWC)
// raw2 : overlays hp1 (hp1 dead after conv1)
#define OFF_HP1   0ULL
#define OFF_HP2   71368704ULL
#define OFF_STATS 107053056ULL
#define OFF_XH    107057152ULL
#define OFF_WHP   123834368ULL
#define OFF_W1P   124096512ULL
#define OFF_W2P   126455808ULL
#define OFF_RAW1  127635456ULL
#define OFF_RAW2  0ULL
#define ZERO_V16  6691072ULL   // (hp1+hp2+stats) bytes / 16

// ---------------- zero fill (borders of hp1/hp2 + stats) ----------------
__global__ __launch_bounds__(256) void zero_ws_kernel(float4* __restrict__ p, size_t n)
{
    size_t i = (size_t)blockIdx.x * blockDim.x + threadIdx.x;
    size_t stride = (size_t)gridDim.x * blockDim.x;
    for (; i < n; i += stride) p[i] = make_float4(0.f, 0.f, 0.f, 0.f);
}

// ---------------- NCHW f32 -> NHWC(-padded) f16 transpose ----------------
__global__ __launch_bounds__(256) void nchw2nhwc_kernel(
    const float* __restrict__ src, _Float16* __restrict__ dst,
    int C, int H, int W, int dH, int dW, int y0, int x0, int dstC)
{
    __shared__ float tile[32][33];
    const int tx = threadIdx.x, ty = threadIdx.y;
    const int nCT = C >> 5, nXT = W >> 5;
    int blk = blockIdx.x;
    const int ct = blk % nCT; blk /= nCT;
    const int xt = blk % nXT; blk /= nXT;
    const int y  = blk % H;
    const int b  = blk / H;
#pragma unroll
    for (int i = 0; i < 4; ++i) {
        const int c = ct * 32 + ty * 4 + i;
        tile[ty * 4 + i][tx] = src[(((size_t)b * C + c) * H + y) * W + xt * 32 + tx];
    }
    __syncthreads();
#pragma unroll
    for (int i = 0; i < 4; ++i) {
        const int xx = xt * 32 + ty * 4 + i;
        dst[(((size_t)b * dH + y0 + y) * dW + (x0 + xx)) * dstC + ct * 32 + tx] =
            (_Float16)tile[tx][ty * 4 + i];
    }
}

// ---------------- weight pack: OIHW f32 -> WMMA A-fragment layout f16 ----
// dst[((t*16 + ocTile)*32 + lane)*16 + h], k = tap*Cin + ic
__global__ __launch_bounds__(256) void pack_wmma_a_kernel(
    const float* __restrict__ src, _Float16* __restrict__ dst,
    int KT, int cinLog2, int ntaps)
{
    const int total = KT << 13;
    const int e = blockIdx.x * 256 + threadIdx.x;
    if (e >= total) return;
    const int h      = e & 15;
    const int lane   = (e >> 4) & 31;
    const int ocTile = (e >> 9) & 15;
    const int t      = e >> 13;
    const int lane15 = lane & 15, hi = lane >> 4;
    const int kk = h + hi * 16;   // contiguous k order: k = 16*half + s
    const int k  = t * 32 + kk;
    const int oc = ocTile * 16 + lane15;
    const int tap = k >> cinLog2;
    const int ic  = k & ((1 << cinLog2) - 1);
    dst[e] = (_Float16)src[((size_t)oc * (1 << cinLog2) + ic) * ntaps + tap];
}

// ---------------- conv1x1 (GEMM 256x512 * 512xN) + fused Haar IDWT -------
__global__ __launch_bounds__(256) void conv1x1_haar_wmma_kernel(
    const _Float16* __restrict__ xh,   // [16][1024][512]
    const _Float16* __restrict__ Ap,   // packed [16][16][32][16]
    const float* __restrict__ LH, const float* __restrict__ HL,
    const float* __restrict__ HH, const float* __restrict__ bh,
    _Float16* __restrict__ hp1)        // [16][66][66][512]
{
    const int lane = threadIdx.x, wv = threadIdx.y;
    const int b  = blockIdx.x >> 4;
    const int n0 = (blockIdx.x & 15) * 64;
    const int oc0 = (wv & 3) * 64;
    const int nW  = n0 + (wv >> 2) * 32;
    const int lane15 = lane & 15, hi = lane >> 4;

    v8f acc[4][2];
#pragma unroll
    for (int q = 0; q < 4; ++q)
#pragma unroll
        for (int p = 0; p < 2; ++p) acc[q][p] = (v8f)0.0f;

    for (int t = 0; t < 16; ++t) {
        const int ic0 = t * 32;
        const _Float16* bp = xh + ((size_t)b * 1024 + nW + lane15) * 512 + ic0 + hi * 16;
        const v16h bf0 = *(const v16h*)bp;
        const v16h bf1 = *(const v16h*)(bp + (size_t)16 * 512);
        const _Float16* apl = Ap + (((size_t)t * 16 + (oc0 >> 4)) * 32 + lane) * 16;
        __builtin_prefetch(apl + 8192, 0, 3);
#pragma unroll
        for (int q = 0; q < 4; ++q) {
            const v16h af = *(const v16h*)(apl + (size_t)q * 512);
            acc[q][0] = __builtin_amdgcn_wmma_f32_16x16x32_f16(
                false, af, false, bf0, (short)0, acc[q][0], false, false);
            acc[q][1] = __builtin_amdgcn_wmma_f32_16x16x32_f16(
                false, af, false, bf1, (short)0, acc[q][1], false, false);
        }
    }

    // Haar IDWT epilogue: each lane holds 8 consecutive oc for one pixel
#pragma unroll
    for (int q = 0; q < 4; ++q) {
        const int ocB = oc0 + q * 16 + hi * 8;
#pragma unroll
        for (int p = 0; p < 2; ++p) {
            const int n  = nW + p * 16 + lane15;
            const int yy = n >> 5, xx = n & 31;
            const size_t cb = ((size_t)b * 256 + ocB) * 1024 + n;
            v8h tl, tr, bl, br;
#pragma unroll
            for (int r = 0; r < 8; ++r) {
                const float ll = acc[q][p][r] + bh[ocB + r];
                const float lh = LH[cb + (size_t)r * 1024];
                const float hl = HL[cb + (size_t)r * 1024];
                const float hh = HH[cb + (size_t)r * 1024];
                tl[r] = (_Float16)(0.5f * (ll - lh - hl + hh));
                tr[r] = (_Float16)(0.5f * (ll - lh + hl - hh));
                bl[r] = (_Float16)(0.5f * (ll + lh - hl - hh));
                br[r] = (_Float16)(0.5f * (ll + lh + hl + hh));
            }
            const size_t base =
                (((size_t)b * 66 + 1 + 2 * yy) * 66 + (1 + 2 * xx)) * 512 + 256 + ocB;
            *(v8h*)(hp1 + base)                 = tl;
            *(v8h*)(hp1 + base + 512)           = tr;
            *(v8h*)(hp1 + base + 66 * 512)      = bl;
            *(v8h*)(hp1 + base + 66 * 512 + 512) = br;
        }
    }
}

// ---------------- implicit-GEMM 3x3 conv via WMMA ------------------------
// hp: padded NHWC f16 [16][66][66][CIN]; Ap packed weights; raw: NHWC f16 [16][64][64][256]
template<int CIN, int TPT, int KT>
__global__ __launch_bounds__(256) void conv3x3_wmma_kernel(
    const _Float16* __restrict__ hp,
    const _Float16* __restrict__ Ap,
    _Float16* __restrict__ raw)
{
    const int lane = threadIdx.x, wv = threadIdx.y;
    const int b = blockIdx.x >> 6;
    const int y = blockIdx.x & 63;
    const int oc0 = (wv & 3) * 64;
    const int x0  = (wv >> 2) * 32;
    const int lane15 = lane & 15, hi = lane >> 4;

    v8f acc[4][2];
#pragma unroll
    for (int q = 0; q < 4; ++q)
#pragma unroll
        for (int p = 0; p < 2; ++p) acc[q][p] = (v8f)0.0f;

    for (int t = 0; t < KT; ++t) {
        const int tap = t / TPT;
        const int ic0 = (t - tap * TPT) * 32;
        const int dy = tap / 3;
        const int dx = tap - dy * 3;
        const _Float16* bp =
            hp + (((size_t)b * 66 + (y + dy)) * 66 + (x0 + lane15 + dx)) * CIN + ic0 + hi * 16;
        const v16h bf0 = *(const v16h*)bp;
        const v16h bf1 = *(const v16h*)(bp + (size_t)16 * CIN);
        const _Float16* apl = Ap + (((size_t)t * 16 + (oc0 >> 4)) * 32 + lane) * 16;
        __builtin_prefetch(apl + 8192, 0, 3);
#pragma unroll
        for (int q = 0; q < 4; ++q) {
            const v16h af = *(const v16h*)(apl + (size_t)q * 512);
            acc[q][0] = __builtin_amdgcn_wmma_f32_16x16x32_f16(
                false, af, false, bf0, (short)0, acc[q][0], false, false);
            acc[q][1] = __builtin_amdgcn_wmma_f32_16x16x32_f16(
                false, af, false, bf1, (short)0, acc[q][1], false, false);
        }
    }

    const size_t outRow = ((size_t)b * 64 + y) * 64;
#pragma unroll
    for (int q = 0; q < 4; ++q) {
        const int ocB = oc0 + q * 16 + hi * 8;
#pragma unroll
        for (int p = 0; p < 2; ++p) {
            const int px = x0 + p * 16 + lane15;
            v8h o;
#pragma unroll
            for (int r = 0; r < 8; ++r) o[r] = (_Float16)acc[q][p][r];
            *(v8h*)(raw + (outRow + px) * 256 + ocB) = o;
        }
    }
}

// ---------------- per-channel batch stats (sum, sumsq) -------------------
__global__ __launch_bounds__(256) void chan_stats_kernel(
    const _Float16* __restrict__ raw, float* __restrict__ sum, float* __restrict__ sq)
{
    const int c = threadIdx.x;
    const size_t p0 = (size_t)blockIdx.x * 256;
    float s = 0.f, s2 = 0.f;
    for (int i = 0; i < 256; ++i) {
        const float v = (float)raw[(p0 + i) * 256 + c];
        s += v; s2 += v * v;
    }
    atomicAdd(&sum[c], s);
    atomicAdd(&sq[c], s2);
}

// ---------------- BN+ReLU, write into padded NHWC f16 --------------------
__global__ __launch_bounds__(256) void bn_relu_mid_kernel(
    const _Float16* __restrict__ raw, const float* __restrict__ sum,
    const float* __restrict__ sq, const float* __restrict__ g,
    const float* __restrict__ be, _Float16* __restrict__ hp2)
{
    const size_t e = (size_t)blockIdx.x * 256 + threadIdx.x;  // 16.7M elements
    const int c = (int)(e & 255);
    const size_t pix = e >> 8;
    const int x = (int)(pix & 63), y = (int)((pix >> 6) & 63), b = (int)(pix >> 12);
    const float inv = 1.f / 65536.f;
    const float m  = sum[c] * inv;
    const float var = sq[c] * inv - m * m;
    const float sc = g[c] * rsqrtf(var + 1e-5f);
    const float sh = be[c] - m * sc;
    float v = (float)raw[e] * sc + sh;
    v = v > 0.f ? v : 0.f;
    hp2[(((size_t)b * 66 + 1 + y) * 66 + 1 + x) * 256 + c] = (_Float16)v;
}

// ---------------- BN+ReLU + NHWC->NCHW transpose, fp32 out ---------------
__global__ __launch_bounds__(256) void bn_relu_final_kernel(
    const _Float16* __restrict__ raw, const float* __restrict__ sum,
    const float* __restrict__ sq, const float* __restrict__ g,
    const float* __restrict__ be, float* __restrict__ out)
{
    __shared__ float tile[32][33];
    const int tx = threadIdx.x, ty = threadIdx.y;
    int blk = blockIdx.x;
    const int ct = blk & 7;  blk >>= 3;
    const int xt = blk & 1;  blk >>= 1;
    const int y  = blk & 63;
    const int b  = blk >> 6;
#pragma unroll
    for (int i = 0; i < 4; ++i) {
        const int xx = xt * 32 + ty * 4 + i;
        tile[ty * 4 + i][tx] =
            (float)raw[(((size_t)b * 64 + y) * 64 + xx) * 256 + ct * 32 + tx];
    }
    __syncthreads();
    const float inv = 1.f / 65536.f;
#pragma unroll
    for (int i = 0; i < 4; ++i) {
        const int c = ct * 32 + ty * 4 + i;
        const float m  = sum[c] * inv;
        const float var = sq[c] * inv - m * m;
        const float sc = g[c] * rsqrtf(var + 1e-5f);
        const float sh = be[c] - m * sc;
        float v = tile[tx][ty * 4 + i] * sc + sh;
        v = v > 0.f ? v : 0.f;
        out[(((size_t)b * 256 + c) * 64 + y) * 64 + xt * 32 + tx] = v;
    }
}

extern "C" void kernel_launch(void* const* d_in, const int* in_sizes, int n_in,
                              void* d_out, int out_size, void* d_ws, size_t ws_size,
                              hipStream_t stream)
{
    (void)in_sizes; (void)n_in; (void)out_size; (void)ws_size;
    const float* x    = (const float*)d_in[0];
    const float* skip = (const float*)d_in[1];
    const float* LH   = (const float*)d_in[2];
    const float* HL   = (const float*)d_in[3];
    const float* HH   = (const float*)d_in[4];
    const float* Wh   = (const float*)d_in[5];
    const float* bh   = (const float*)d_in[6];
    const float* W1   = (const float*)d_in[7];
    const float* g1   = (const float*)d_in[8];
    const float* be1  = (const float*)d_in[9];
    const float* W2   = (const float*)d_in[10];
    const float* g2   = (const float*)d_in[11];
    const float* be2  = (const float*)d_in[12];

    char* ws = (char*)d_ws;
    _Float16* hp1  = (_Float16*)(ws + OFF_HP1);
    _Float16* hp2  = (_Float16*)(ws + OFF_HP2);
    float*    stats = (float*)(ws + OFF_STATS);
    _Float16* xh   = (_Float16*)(ws + OFF_XH);
    _Float16* Whp  = (_Float16*)(ws + OFF_WHP);
    _Float16* W1p  = (_Float16*)(ws + OFF_W1P);
    _Float16* W2p  = (_Float16*)(ws + OFF_W2P);
    _Float16* raw1 = (_Float16*)(ws + OFF_RAW1);
    _Float16* raw2 = (_Float16*)(ws + OFF_RAW2);
    float* sum1 = stats;       float* sq1 = stats + 256;
    float* sum2 = stats + 512; float* sq2 = stats + 768;

    const dim3 blk32x8(32, 8);

    // 1. zero hp1 + hp2 + stats (borders must be 0; stats accumulated atomically)
    zero_ws_kernel<<<2048, 256, 0, stream>>>((float4*)ws, ZERO_V16);

    // 2. x [16,512,32,32] -> xh NHWC f16
    nchw2nhwc_kernel<<<8192, blk32x8, 0, stream>>>(x, xh, 512, 32, 32, 32, 32, 0, 0, 512);

    // 3. skip [16,256,64,64] -> hp1 channels 0..255 (padded)
    nchw2nhwc_kernel<<<16384, blk32x8, 0, stream>>>(skip, hp1, 256, 64, 64, 66, 66, 1, 1, 512);

    // 4. pack weights into WMMA A-fragment layout
    pack_wmma_a_kernel<<<(16  * 8192 + 255) / 256, 256, 0, stream>>>(Wh, Whp, 16,  9, 1);
    pack_wmma_a_kernel<<<(144 * 8192 + 255) / 256, 256, 0, stream>>>(W1, W1p, 144, 9, 9);
    pack_wmma_a_kernel<<<(72  * 8192 + 255) / 256, 256, 0, stream>>>(W2, W2p, 72,  8, 9);

    // 5. conv1x1 + Haar IDWT -> hp1 channels 256..511
    conv1x1_haar_wmma_kernel<<<256, blk32x8, 0, stream>>>(xh, Whp, LH, HL, HH, bh, hp1);

    // 6. conv3x3 #1 (512 -> 256), implicit GEMM, K = 4608
    conv3x3_wmma_kernel<512, 16, 144><<<1024, blk32x8, 0, stream>>>(hp1, W1p, raw1);

    // 7-8. batch stats + BN/ReLU -> hp2 (padded NHWC f16)
    chan_stats_kernel<<<256, 256, 0, stream>>>(raw1, sum1, sq1);
    bn_relu_mid_kernel<<<65536, 256, 0, stream>>>(raw1, sum1, sq1, g1, be1, hp2);

    // 9. conv3x3 #2 (256 -> 256), K = 2304
    conv3x3_wmma_kernel<256, 8, 72><<<1024, blk32x8, 0, stream>>>(hp2, W2p, raw2);

    // 10-11. batch stats + BN/ReLU + NCHW transpose -> d_out fp32
    chan_stats_kernel<<<256, 256, 0, stream>>>(raw2, sum2, sq2);
    bn_relu_final_kernel<<<16384, blk32x8, 0, stream>>>(raw2, sum2, sq2, g2, be2,
                                                        (float*)d_out);
}
